// Seq2SeqDecoder_7035156431439
// MI455X (gfx1250) — matrix operand order repaired and checked by reading, not compiled, not run on hardware
//
#include <hip/hip_runtime.h>
#include <hip/hip_bf16.h>

// Problem dims
#define B_  32
#define T_  64
#define S_  256
#define E_  512
#define H_  512
#define C_  32000
#define K_  1024      // E+H for gates GEMM, 2H for projection GEMM (both 1024)

typedef __attribute__((ext_vector_type(16))) __bf16 bf16x16;
typedef __attribute__((ext_vector_type(8)))  __bf16 bf16x8;
typedef __attribute__((ext_vector_type(8)))  float  f32x8;

// ---------------------------------------------------------------------------
// WMMA fragment loaders (CDNA5 ISA layouts, wave32)
// A (16-bit, 16x32 MxK): lane m<16 holds row m, K {kb+0..7, kb+16..23};
//                        lane m+16 holds row m, K {kb+8..15, kb+24..31}.
// B (16-bit, 32x16 KxN): lane n<16 holds col n, K {kb+0..7, kb+16..23};
//                        lane n+16 holds col n, K {kb+8..15, kb+24..31}.
// Weights are stored row-major W[N,K] (= B^T), so "col n" = weight row n.
// ---------------------------------------------------------------------------
__device__ __forceinline__ bf16x16 load_a_frag(const __bf16* rowBase, int lane) {
  const int k0 = (lane & 16) ? 8 : 0;
  bf16x8 lo = *reinterpret_cast<const bf16x8*>(rowBase + k0);
  bf16x8 hi = *reinterpret_cast<const bf16x8*>(rowBase + k0 + 16);
  bf16x16 v;
#pragma unroll
  for (int i = 0; i < 8; ++i) { v[i] = lo[i]; v[i + 8] = hi[i]; }
  return v;
}

__device__ __forceinline__ bf16x16 load_b_frag(const __bf16* rowBase, int lane) {
  const int k0 = (lane & 16) ? 8 : 0;
  bf16x8 lo = *reinterpret_cast<const bf16x8*>(rowBase + k0);
  bf16x8 hi = *reinterpret_cast<const bf16x8*>(rowBase + k0 + 16);
  bf16x16 v;
#pragma unroll
  for (int i = 0; i < 8; ++i) { v[i] = lo[i]; v[i + 8] = hi[i]; }
  return v;
}

#define WMMA_BF16(a, b, acc) \
  __builtin_amdgcn_wmma_f32_16x16x32_bf16(false, (a), false, (b), (short)0, (acc), false, false)

// ---------------------------------------------------------------------------
// Generic bf16 WMMA GEMM: out[row, col] = A[row,:K] . Bw[col,:K] + bias
// A: [32, 1024] bf16 row-major. Bw: [N, 1024] bf16 row-major.
// Block = 64 threads (2 waves); wave w handles M-tile w (rows 16w..16w+15).
// Each wave owns 4 consecutive N-tiles -> gridDim.x = N/64.
// Two-stage ping-pong software pipeline: set S1 is loaded for kt+1 while the
// WMMAs for kt fire from set S0, then S0 reloads for kt+2 while S1's WMMAs
// fire. No loop-carried register rotation -> no v_mov copies / hazard NOPs,
// and v_wmma only waits on loads issued a full stage earlier.
// out index = row*rowStride + rowOff + col (folds the [B,T,C] layout).
// ---------------------------------------------------------------------------
__global__ __launch_bounds__(64) void wmma_gemm_bf16_kernel(
    const __bf16* __restrict__ A, const __bf16* __restrict__ Bw,
    const float* __restrict__ bias0, const float* __restrict__ bias1,
    float* __restrict__ out, long rowStride, long rowOff)
{
  const int lane  = threadIdx.x & 31;
  const int mtile = threadIdx.x >> 5;          // 0..1
  const int r     = lane & 15;
  const long ntBase = (long)blockIdx.x * 4;

  const __bf16* aRow  = A + (size_t)(mtile * 16 + r) * K_;
  const __bf16* bRow0 = Bw + (size_t)((ntBase + 0) * 16 + r) * K_;
  const __bf16* bRow1 = Bw + (size_t)((ntBase + 1) * 16 + r) * K_;
  const __bf16* bRow2 = Bw + (size_t)((ntBase + 2) * 16 + r) * K_;
  const __bf16* bRow3 = Bw + (size_t)((ntBase + 3) * 16 + r) * K_;

  f32x8 acc0, acc1, acc2, acc3;
#pragma unroll
  for (int v = 0; v < 8; ++v) { acc0[v] = 0.f; acc1[v] = 0.f; acc2[v] = 0.f; acc3[v] = 0.f; }

  // Prologue: stage S0 holds fragments for kt = 0
  bf16x16 a0  = load_a_frag(aRow, lane);
  bf16x16 b00 = load_b_frag(bRow0, lane);
  bf16x16 b01 = load_b_frag(bRow1, lane);
  bf16x16 b02 = load_b_frag(bRow2, lane);
  bf16x16 b03 = load_b_frag(bRow3, lane);

  // 15 iterations cover kt = 0..29; loads run one stage ahead (to kt = 30).
  for (int kt = 0; kt < K_ / 32 - 2; kt += 2) {
    const int ko1 = (kt + 1) * 32;
    bf16x16 a1  = load_a_frag(aRow  + ko1, lane);
    bf16x16 b10 = load_b_frag(bRow0 + ko1, lane);
    bf16x16 b11 = load_b_frag(bRow1 + ko1, lane);
    bf16x16 b12 = load_b_frag(bRow2 + ko1, lane);
    bf16x16 b13 = load_b_frag(bRow3 + ko1, lane);

    acc0 = WMMA_BF16(a0, b00, acc0);
    acc1 = WMMA_BF16(a0, b01, acc1);
    acc2 = WMMA_BF16(a0, b02, acc2);
    acc3 = WMMA_BF16(a0, b03, acc3);

    const int ko2 = (kt + 2) * 32;
    a0  = load_a_frag(aRow  + ko2, lane);
    b00 = load_b_frag(bRow0 + ko2, lane);
    b01 = load_b_frag(bRow1 + ko2, lane);
    b02 = load_b_frag(bRow2 + ko2, lane);
    b03 = load_b_frag(bRow3 + ko2, lane);

    acc0 = WMMA_BF16(a1, b10, acc0);
    acc1 = WMMA_BF16(a1, b11, acc1);
    acc2 = WMMA_BF16(a1, b12, acc2);
    acc3 = WMMA_BF16(a1, b13, acc3);
  }

  // Epilogue: S0 holds kt = 30; load S1 for kt = 31, then drain both stages.
  {
    const int ko = (K_ / 32 - 1) * 32;
    bf16x16 a1  = load_a_frag(aRow  + ko, lane);
    bf16x16 b10 = load_b_frag(bRow0 + ko, lane);
    bf16x16 b11 = load_b_frag(bRow1 + ko, lane);
    bf16x16 b12 = load_b_frag(bRow2 + ko, lane);
    bf16x16 b13 = load_b_frag(bRow3 + ko, lane);

    acc0 = WMMA_BF16(a0, b00, acc0);
    acc1 = WMMA_BF16(a0, b01, acc1);
    acc2 = WMMA_BF16(a0, b02, acc2);
    acc3 = WMMA_BF16(a0, b03, acc3);

    acc0 = WMMA_BF16(a1, b10, acc0);
    acc1 = WMMA_BF16(a1, b11, acc1);
    acc2 = WMMA_BF16(a1, b12, acc2);
    acc3 = WMMA_BF16(a1, b13, acc3);
  }

  // D layout (32-bit C/D 16x16): lane n holds col n; VGPR v holds row v (lanes
  // 0-15) or row v+8 (lanes 16-31).
  const int mb = (lane & 16) ? 8 : 0;
  const f32x8 accs[4] = {acc0, acc1, acc2, acc3};
#pragma unroll
  for (int i = 0; i < 4; ++i) {
    const long col = (ntBase + i) * 16 + (lane & 15);
    float bs = bias0[col];
    if (bias1) bs += bias1[col];
#pragma unroll
    for (int v = 0; v < 8; ++v) {
      const long row = mtile * 16 + mb + v;
      out[row * rowStride + rowOff + col] = accs[i][v] + bs;
    }
  }
}

// ---------------------------------------------------------------------------
// One-time weight conversions to bf16
// ---------------------------------------------------------------------------
__global__ void cvt_wproj_kernel(const float* __restrict__ w, __bf16* __restrict__ o) {
  size_t i = (size_t)blockIdx.x * blockDim.x + threadIdx.x;   // C_*K_ threads
  o[i] = (__bf16)w[i];
}

// Fused gate weight: Wg[j, 0:512] = W_ih[j,:], Wg[j, 512:1024] = W_hh[j,:]
__global__ void cvt_wgate_kernel(const float* __restrict__ wih,
                                 const float* __restrict__ whh,
                                 __bf16* __restrict__ o) {
  int i = blockIdx.x * blockDim.x + threadIdx.x;              // 2048*1024 threads
  int j = i >> 10, k = i & 1023;
  float v = (k < E_) ? wih[j * E_ + k] : whh[j * H_ + (k - E_)];
  o[i] = (__bf16)v;
}

// ---------------------------------------------------------------------------
// Per-step: build A operand [x_t | h] in bf16 (embedding gather + h cast)
// ---------------------------------------------------------------------------
__global__ void pack_xh_kernel(const int* __restrict__ ids,
                               const float* __restrict__ emb,
                               const float* __restrict__ h,
                               __bf16* __restrict__ xh, int t) {
  int idx = blockIdx.x * blockDim.x + threadIdx.x;            // 32*1024 threads
  int b = idx >> 10, k = idx & 1023;
  float v;
  if (k < E_) {
    int tok = ids[b * T_ + t];
    v = emb[(size_t)tok * E_ + k];
  } else {
    v = h[b * H_ + (k - E_)];
  }
  xh[idx] = (__bf16)v;
}

// ---------------------------------------------------------------------------
// LSTM cell elementwise (gate order i, f, g, o — PyTorch LSTMCell)
// ---------------------------------------------------------------------------
__global__ void lstm_cell_kernel(const float* __restrict__ gates,
                                 float* __restrict__ h, float* __restrict__ c) {
  int idx = blockIdx.x * blockDim.x + threadIdx.x;            // 32*512 threads
  int b = idx >> 9, u = idx & 511;
  const float* g = gates + (size_t)b * 4 * H_;
  float gi = g[u], gf = g[H_ + u], gg = g[2 * H_ + u], go = g[3 * H_ + u];
  float si = 1.0f / (1.0f + __expf(-gi));
  float sf = 1.0f / (1.0f + __expf(-gf));
  float so = 1.0f / (1.0f + __expf(-go));
  float cn = sf * c[idx] + si * tanhf(gg);
  float hn = so * tanhf(cn);
  c[idx] = cn;
  h[idx] = hn;
}

// ---------------------------------------------------------------------------
// Attention: one block per batch row. scores -> softmax -> context.
// Writes att_probs[b,t,:] and packs [h | context] as bf16 for the projection.
// ---------------------------------------------------------------------------
__global__ __launch_bounds__(256) void attention_kernel(
    const float* __restrict__ enc,   // [B, S, H]
    const float* __restrict__ h,     // [B, H]
    float* __restrict__ att_out,     // [B, T, S]
    __bf16* __restrict__ hc_bf,      // [B, 1024]
    int t)
{
  __shared__ float sh_h[H_];
  __shared__ float sh_p[S_];
  __shared__ float sh_r[S_];
  const int b = blockIdx.x;
  const int tid = threadIdx.x;      // 256 threads

  sh_h[tid]        = h[b * H_ + tid];
  sh_h[tid + 256]  = h[b * H_ + 256 + tid];
  __syncthreads();

  const float* encb = enc + (size_t)b * S_ * H_;

  // score[tid] = dot(h, enc[b, tid, :])
  float score = 0.0f;
  const float4* r4 = reinterpret_cast<const float4*>(encb + (size_t)tid * H_);
  for (int k = 0; k < H_ / 4; ++k) {
    float4 v = r4[k];
    score = fmaf(sh_h[4 * k + 0], v.x, score);
    score = fmaf(sh_h[4 * k + 1], v.y, score);
    score = fmaf(sh_h[4 * k + 2], v.z, score);
    score = fmaf(sh_h[4 * k + 3], v.w, score);
  }

  // softmax over S=256
  sh_r[tid] = score;
  __syncthreads();
  for (int o = 128; o > 0; o >>= 1) {
    if (tid < o) sh_r[tid] = fmaxf(sh_r[tid], sh_r[tid + o]);
    __syncthreads();
  }
  const float mx = sh_r[0];
  __syncthreads();
  float p = __expf(score - mx);
  sh_r[tid] = p;
  __syncthreads();
  for (int o = 128; o > 0; o >>= 1) {
    if (tid < o) sh_r[tid] += sh_r[tid + o];
    __syncthreads();
  }
  p *= (1.0f / sh_r[0]);
  sh_p[tid] = p;
  att_out[((size_t)b * T_ + t) * S_ + tid] = p;
  __syncthreads();

  // context[k] = sum_s p[s] * enc[b, s, k]   (coalesced over k)
  float c0 = 0.0f, c1 = 0.0f;
  for (int s = 0; s < S_; ++s) {
    const float a = sh_p[s];
    const float* row = encb + (size_t)s * H_;
    c0 = fmaf(a, row[tid], c0);
    c1 = fmaf(a, row[tid + 256], c1);
  }
  __bf16* hc = hc_bf + (size_t)b * 1024;
  hc[tid]             = (__bf16)sh_h[tid];
  hc[tid + 256]       = (__bf16)sh_h[tid + 256];
  hc[512 + tid]       = (__bf16)c0;
  hc[512 + tid + 256] = (__bf16)c1;
}

// ---------------------------------------------------------------------------
// Host side
// ---------------------------------------------------------------------------
extern "C" void kernel_launch(void* const* d_in, const int* in_sizes, int n_in,
                              void* d_out, int out_size, void* d_ws, size_t ws_size,
                              hipStream_t stream)
{
  const int*   ids    = (const int*)  d_in[0];
  const float* enc    = (const float*)d_in[1];
  const float* emb    = (const float*)d_in[2];
  const float* W_ih   = (const float*)d_in[3];
  const float* W_hh   = (const float*)d_in[4];
  const float* b_ih   = (const float*)d_in[5];
  const float* b_hh   = (const float*)d_in[6];
  const float* W_proj = (const float*)d_in[7];
  const float* b_proj = (const float*)d_in[8];

  float* att_out = (float*)d_out;                      // [B,T,S]
  float* log_out = att_out + (size_t)B_ * T_ * S_;     // [B,T,C]

  // Workspace layout (all regions 256B-aligned sizes)
  char* ws = (char*)d_ws;
  const size_t WP_BYTES = (size_t)C_ * K_ * 2;         // 65,536,000
  const size_t WG_BYTES = (size_t)4 * H_ * K_ * 2;     //  4,194,304
  const size_t XH_BYTES = (size_t)B_ * K_ * 2;         //     65,536
  __bf16* Wp_bf = (__bf16*)(ws);
  __bf16* Wg_bf = (__bf16*)(ws + WP_BYTES);
  __bf16* xh_bf = (__bf16*)(ws + WP_BYTES + WG_BYTES);
  __bf16* hc_bf = (__bf16*)(ws + WP_BYTES + WG_BYTES + XH_BYTES);
  float*  h     = (float*) (ws + WP_BYTES + WG_BYTES + 2 * XH_BYTES);
  float*  c     = h + (size_t)B_ * H_;
  float*  gates = c + (size_t)B_ * H_;                 // [32, 2048] fp32

  // One-time: bf16 weight images (W_proj bf16 = 64 MB -> L2-resident)
  cvt_wproj_kernel<<<dim3(((size_t)C_ * K_) / 256), 256, 0, stream>>>(W_proj, Wp_bf);
  cvt_wgate_kernel<<<dim3((4 * H_ * K_) / 256), 256, 0, stream>>>(W_ih, W_hh, Wg_bf);

  // h and c are contiguous: zero both
  hipMemsetAsync(h, 0, (size_t)2 * B_ * H_ * sizeof(float), stream);

  for (int t = 0; t < T_; ++t) {
    // A operand for gates GEMM: [x_t | h] bf16
    pack_xh_kernel<<<dim3((B_ * K_) / 256), 256, 0, stream>>>(ids, emb, h, xh_bf, t);

    // gates[32, 2048] = [x|h] . [W_ih|W_hh]^T + b_ih + b_hh  (N=2048 -> 32 blocks)
    wmma_gemm_bf16_kernel<<<dim3((4 * H_) / 64), 64, 0, stream>>>(
        xh_bf, Wg_bf, b_ih, b_hh, gates, (long)(4 * H_), 0L);

    lstm_cell_kernel<<<dim3((B_ * H_) / 256), 256, 0, stream>>>(gates, h, c);

    attention_kernel<<<dim3(B_), 256, 0, stream>>>(enc, h, att_out, hc_bf, t);

    // logits[b, t, :] = [h|ctx] . W_proj^T + b_proj  (N=32000 -> 500 blocks)
    wmma_gemm_bf16_kernel<<<dim3(C_ / 64), 64, 0, stream>>>(
        hc_bf, Wp_bf, b_proj, nullptr, log_out,
        (long)T_ * C_, (long)t * C_);
  }
}
